// MultiHeadedAttention_2551210574136
// MI455X (gfx1250) — compile-verified, matrix operand order checked by reading, not run on hardware
//
#include <hip/hip_runtime.h>
#include <hip/hip_fp16.h>

// Problem constants (fixed by the reference)
#define AB 2        // batch
#define AS 2048     // sequence
#define AD 1024     // d_model
#define AH 16       // heads
#define ADK 64      // d_k
#define ABS (AB*AS) // 4096 rows

typedef __attribute__((ext_vector_type(16))) _Float16 v16h;
typedef __attribute__((ext_vector_type(8)))  _Float16 v8h;
typedef __attribute__((ext_vector_type(8)))  float    v8f;

// Load one 16-element f16 WMMA fragment (A- or B-role) for lane (row = lane&15,
// khalf = lane>>4). Caller passes p = rowBase + k0 + khalf*8; elements e=0..7 map
// to K = k0+khalf*8+e, e=8..15 map to K = k0+16+khalf*8+(e-8)  (ISA 7.12.2 layout).
__device__ __forceinline__ v16h load_frag16(const _Float16* p) {
  v8h lo = *reinterpret_cast<const v8h*>(p);
  v8h hi = *reinterpret_cast<const v8h*>(p + 16);
  v16h r;
#pragma unroll
  for (int i = 0; i < 8; ++i) { r[i] = lo[i]; r[i + 8] = hi[i]; }
  return r;
}

__device__ __forceinline__ v8f wmma_f16(v16h a, v16h b, v8f c) {
  // (neg_a, A, neg_b, B, c_mod, C, reuse_a, reuse_b)
  return __builtin_amdgcn_wmma_f32_16x16x32_f16(false, a, false, b, (short)0, c,
                                                false, false);
}

// ---------------------------------------------------------------- elementwise
__global__ void cvt_f32_f16(const float* __restrict__ in, _Float16* __restrict__ out,
                            int n) {
  int i = blockIdx.x * blockDim.x + threadIdx.x;
  if (i < n) out[i] = (_Float16)in[i];
}

// W[K,N] fp32  ->  Wt[N,K] f16   (K = N = AD)
__global__ void transpose_w(const float* __restrict__ W, _Float16* __restrict__ Wt) {
  int k = blockIdx.x * blockDim.x + threadIdx.x;
  int n = blockIdx.y;
  if (k < AD) Wt[(size_t)n * AD + k] = (_Float16)W[(size_t)k * AD + n];
}

// V[BS, D] f16 -> Vt[(b*H+h)*DK + dk][S] f16 (K-major over keys for PV frags)
__global__ void transpose_v(const _Float16* __restrict__ V,
                            _Float16* __restrict__ Vt) {
  int s = blockIdx.x * blockDim.x + threadIdx.x;
  int r = blockIdx.y;              // (b*H + h)*64 + dk
  int dk = r & (ADK - 1);
  int bh = r >> 6;
  int h  = bh & (AH - 1);
  int b  = bh >> 4;
  if (s < AS)
    Vt[(size_t)r * AS + s] = V[((size_t)b * AS + s) * AD + h * ADK + dk];
}

// ------------------------------------------------------------------- GEMM
// Y[M,N] = X[M,K] * Wt[N,K]^T + bias.
// Register-blocked: one wave computes a 32x64 tile (2 A-frags x 4 B-frags ->
// 8 WMMAs per 12 b128 loads), K-loop step 32.
__global__ __launch_bounds__(256)
void gemm_bias(const _Float16* __restrict__ X, const _Float16* __restrict__ Wt,
               const float* __restrict__ bias, _Float16* __restrict__ Yhf,
               float* __restrict__ Yf, int M, int N, int K) {
  const int lane = threadIdx.x & 31;
  const int wave = threadIdx.x >> 5;
  const int m0 = blockIdx.x * 32;
  const int n0 = (blockIdx.y * 8 + wave) * 64;
  const int row = lane & 15;   // A row / B col for this lane
  const int kh  = lane >> 4;   // K half select

  const _Float16* xr0 = X + (size_t)(m0 + row) * K + kh * 8;
  const _Float16* xr1 = X + (size_t)(m0 + 16 + row) * K + kh * 8;
  const _Float16* wr[4];
#pragma unroll
  for (int t = 0; t < 4; ++t)
    wr[t] = Wt + (size_t)(n0 + t * 16 + row) * K + kh * 8;

  v8f acc[2][4] = {};
  for (int k0 = 0; k0 < K; k0 += 32) {
    __builtin_prefetch(xr0 + k0 + 512, 0, 1);   // global_prefetch_b8
    __builtin_prefetch(xr1 + k0 + 512, 0, 1);
    v16h a0 = load_frag16(xr0 + k0);
    v16h a1 = load_frag16(xr1 + k0);
#pragma unroll
    for (int t = 0; t < 4; ++t) {
      v16h b = load_frag16(wr[t] + k0);
      acc[0][t] = wmma_f16(a0, b, acc[0][t]);
      acc[1][t] = wmma_f16(a1, b, acc[1][t]);
    }
  }
#pragma unroll
  for (int t = 0; t < 4; ++t) {
    const float bv = bias[n0 + t * 16 + row];
#pragma unroll
    for (int mi = 0; mi < 2; ++mi)
#pragma unroll
      for (int r = 0; r < 8; ++r) {
        float v = acc[mi][t][r] + bv;
        size_t off =
            (size_t)(m0 + mi * 16 + r + 8 * kh) * N + n0 + t * 16 + row;
        if (Yhf) Yhf[off] = (_Float16)v;
        else     Yf[off]  = v;
      }
  }
}

// --------------------------------------------------------------- attention
// One wave per (b, h, 16-row query tile). Flash-attention over 32-key tiles:
// 4 QK^T WMMAs + softmax + 4 PV WMMAs per tile. P re-layouted through LDS.
__global__ __launch_bounds__(32)
void flash_attn(const _Float16* __restrict__ Q, const _Float16* __restrict__ Kp,
                const _Float16* __restrict__ Vt, const int* __restrict__ mask,
                _Float16* __restrict__ Ctx) {
  const int lane = threadIdx.x;
  const int qt = blockIdx.x, h = blockIdx.y, b = blockIdx.z;
  const int q0  = qt * 16;
  const int row = lane & 15;
  const int kh  = lane >> 4;

  // Q fragments for K-dim 0..31 and 32..63
  const _Float16* qrow =
      Q + ((size_t)(b * AS + q0 + row) * AD + h * ADK) + kh * 8;
  v16h aq0 = load_frag16(qrow);
  v16h aq1 = load_frag16(qrow + 32);

  v8f acc[4] = {};                 // 16x64 output accumulator (4 n-tiles)
  float run_m[8], run_l[8];
#pragma unroll
  for (int r = 0; r < 8; ++r) { run_m[r] = -INFINITY; run_l[r] = 0.f; }

  __shared__ __align__(16) _Float16 Plds[16 * 32];
  const int* mrow = mask + ((size_t)b * AS + q0) * AS;

  for (int j = 0; j < AS; j += 32) {
    // ---- scores: 16x32 = two 16x16 tiles ----
    v8f s[2];
#pragma unroll
    for (int nt = 0; nt < 2; ++nt) {
      const _Float16* krow =
          Kp + ((size_t)(b * AS + j + nt * 16 + row) * AD + h * ADK) + kh * 8;
      v8f c = {};
      c = wmma_f16(aq0, load_frag16(krow), c);
      c = wmma_f16(aq1, load_frag16(krow + 32), c);
      s[nt] = c;
    }
    // ---- scale + mask ----
#pragma unroll
    for (int nt = 0; nt < 2; ++nt)
#pragma unroll
      for (int r = 0; r < 8; ++r) {
        int m   = r + 8 * kh;
        int key = j + nt * 16 + row;
        float v = s[nt][r] * 0.125f;             // 1/sqrt(64)
        if (mrow[(size_t)m * AS + key] == 0) v = -1e9f;
        s[nt][r] = v;
      }
    // ---- online softmax (row reductions across 16-lane half-groups) ----
    float sf[8];
#pragma unroll
    for (int r = 0; r < 8; ++r) {
      float mx = fmaxf(s[0][r], s[1][r]);
      for (int d = 1; d < 16; d <<= 1) mx = fmaxf(mx, __shfl_xor(mx, d, 32));
      float nm = fmaxf(run_m[r], mx);
      float f  = __expf(run_m[r] - nm);
      float p0 = __expf(s[0][r] - nm);
      float p1 = __expf(s[1][r] - nm);
      s[0][r] = p0; s[1][r] = p1;
      float sum = p0 + p1;
      for (int d = 1; d < 16; d <<= 1) sum += __shfl_xor(sum, d, 32);
      run_l[r] = run_l[r] * f + sum;
      run_m[r] = nm;
      sf[r] = f;
    }
#pragma unroll
    for (int t = 0; t < 4; ++t)
#pragma unroll
      for (int r = 0; r < 8; ++r) acc[t][r] *= sf[r];

    // ---- P: C/D layout -> A-fragment layout via LDS ----
    __syncthreads();
#pragma unroll
    for (int r = 0; r < 8; ++r) {
      int m = r + 8 * kh;
      Plds[m * 32 + row]      = (_Float16)s[0][r];
      Plds[m * 32 + 16 + row] = (_Float16)s[1][r];
    }
    __syncthreads();
    v16h pa = load_frag16(&Plds[row * 32 + kh * 8]);

    // ---- PV: 16x32 @ 32x64 ----
#pragma unroll
    for (int t = 0; t < 4; ++t) {
      const _Float16* vrow =
          Vt + ((size_t)((b * AH + h) * ADK + t * 16 + row)) * AS + j + kh * 8;
      acc[t] = wmma_f16(pa, load_frag16(vrow), acc[t]);
    }
    __syncthreads();
  }

  // ---- epilogue: normalize and emit context (f16, [BS, D]) ----
#pragma unroll
  for (int t = 0; t < 4; ++t)
#pragma unroll
    for (int r = 0; r < 8; ++r) {
      int m = r + 8 * kh;
      float v = acc[t][r] / run_l[r];
      Ctx[((size_t)(b * AS + q0 + m)) * AD + h * ADK + t * 16 + row] =
          (_Float16)v;
    }
}

// ------------------------------------------------------------------- launch
extern "C" void kernel_launch(void* const* d_in, const int* in_sizes, int n_in,
                              void* d_out, int out_size, void* d_ws, size_t ws_size,
                              hipStream_t stream) {
  const float* query = (const float*)d_in[0];
  const float* key_  = (const float*)d_in[1];
  const float* value = (const float*)d_in[2];
  const int*   mask  = (const int*)d_in[3];
  const float* Wq = (const float*)d_in[4];  const float* bq = (const float*)d_in[5];
  const float* Wk = (const float*)d_in[6];  const float* bk = (const float*)d_in[7];
  const float* Wv = (const float*)d_in[8];  const float* bv = (const float*)d_in[9];
  const float* Wo = (const float*)d_in[10]; const float* bo = (const float*)d_in[11];
  float* out = (float*)d_out;

  const size_t nBD = (size_t)ABS * AD;   // 4096*1024
  const size_t nDD = (size_t)AD * AD;

  char* w = (char*)d_ws;
  size_t off = 0;
  auto grab = [&](size_t elems) {
    _Float16* p = (_Float16*)(w + off);
    off += ((elems * sizeof(_Float16)) + 255) & ~(size_t)255;
    return p;
  };
  _Float16* Xq  = grab(nBD);
  _Float16* Xk  = grab(nBD);
  _Float16* Xv  = grab(nBD);
  _Float16* Wtq = grab(nDD);
  _Float16* Wtk = grab(nDD);
  _Float16* Wtv = grab(nDD);
  _Float16* Wto = grab(nDD);
  _Float16* Qb  = grab(nBD);
  _Float16* Kb  = grab(nBD);
  _Float16* Vb  = grab(nBD);
  _Float16* Vt  = grab(nBD);
  _Float16* Ctx = Xq;   // Xq is dead after the Q projection; reuse for context

  // 1) fp32 -> f16 activation staging
  {
    int n = (int)nBD;
    dim3 g((n + 255) / 256), blk(256);
    cvt_f32_f16<<<g, blk, 0, stream>>>(query, Xq, n);
    cvt_f32_f16<<<g, blk, 0, stream>>>(key_,  Xk, n);
    cvt_f32_f16<<<g, blk, 0, stream>>>(value, Xv, n);
  }
  // 2) weight transposes (K-major f16)
  {
    dim3 g(AD / 256, AD), blk(256);
    transpose_w<<<g, blk, 0, stream>>>(Wq, Wtq);
    transpose_w<<<g, blk, 0, stream>>>(Wk, Wtk);
    transpose_w<<<g, blk, 0, stream>>>(Wv, Wtv);
    transpose_w<<<g, blk, 0, stream>>>(Wo, Wto);
  }
  // 3) Q/K/V projections (WMMA f16, fp32 accumulate)
  {
    dim3 g(ABS / 32, (AD / 64) / 8), blk(256);
    gemm_bias<<<g, blk, 0, stream>>>(Xq, Wtq, bq, Qb, nullptr, ABS, AD, AD);
    gemm_bias<<<g, blk, 0, stream>>>(Xk, Wtk, bk, Kb, nullptr, ABS, AD, AD);
    gemm_bias<<<g, blk, 0, stream>>>(Xv, Wtv, bv, Vb, nullptr, ABS, AD, AD);
  }
  // 4) V -> Vt (keys contiguous for PV B-fragments)
  {
    dim3 g(AS / 256, AB * AD), blk(256);
    transpose_v<<<g, blk, 0, stream>>>(Vb, Vt);
  }
  // 5) flash attention
  {
    dim3 g(AS / 16, AH, AB), blk(32);
    flash_attn<<<g, blk, 0, stream>>>(Qb, Kb, Vt, mask, Ctx);
  }
  // 6) output projection, fp32 result to d_out
  {
    dim3 g(ABS / 32, (AD / 64) / 8), blk(256);
    gemm_bias<<<g, blk, 0, stream>>>(Ctx, Wto, bo, nullptr, out, ABS, AD, AD);
  }
  (void)in_sizes; (void)n_in; (void)out_size; (void)ws_size;
}
